// robust_decomp_enc_87986700026366
// MI455X (gfx1250) — compile-verified, matrix operand order checked by reading, not run on hardware
//
#include <hip/hip_runtime.h>
#include <hip/hip_bf16.h>

// ---------------- types ----------------
typedef __bf16 bf16;
typedef __attribute__((ext_vector_type(16))) __bf16 v16bf;
typedef __attribute__((ext_vector_type(8)))  float  v8f;
typedef int v4i_gcc __attribute__((vector_size(16)));   // matches builtin prototype

// ---------------- problem constants ----------------
#define BATCH   8
#define LEN     336
#define EMB     512
#define NC      4096        // B*E columns
#define NDIM    335         // L-1
#define MROWS   980         // 3L-28
#define HM      311         // L-25
#define KP      992         // MROWS padded to 31*32 (reduction dim)
#define MP      1024        // MROWS padded to 8*128 (output-row tiling)
#define SP      384         // NDIM padded to 3*128 (final gemm rows)
#define OUTHALF 1376256     // 8*336*512

// ---------------- WMMA GEMM tiling ----------------
#define BM 128
#define BN 128
#define BK 32
#define KSTEPS (KP / BK)    // 31
#define LDSS 40             // padded LDS row stride in bf16 (80B, 16B aligned, bank-spread)

#define USE_ASYNC_LDS 1     // CDNA5 GLOBAL_LOAD_ASYNC_TO_LDS for the A tile

union FragU { uint4 q[2]; v16bf v; };

// Per-lane bf16 fragment load following CDNA5 16-bit A layout:
// lane r=lane&15 selects row; half=lane>>4 selects K groups {8h..8h+7, 16+8h..16+8h+7}
__device__ __forceinline__ v16bf load_frag(const bf16* base, int rowbase, int lane) {
    int r = lane & 15, half = lane >> 4;
    const bf16* p = base + (rowbase + r) * LDSS;
    FragU f;
    f.q[0] = *(const uint4*)(p + 8 * half);
    f.q[1] = *(const uint4*)(p + 16 + 8 * half);
    return f.v;
}

// 16B global -> LDS async copy (ASYNCcnt-tracked), per-lane addresses.
__device__ __forceinline__ void async_copy_b128(void* lds_ptr, const void* gptr) {
#if defined(__has_builtin) && __has_builtin(__builtin_amdgcn_global_load_async_to_lds_b128)
    __builtin_amdgcn_global_load_async_to_lds_b128(
        (__attribute__((address_space(1))) v4i_gcc*)(uintptr_t)gptr,
        (__attribute__((address_space(3))) v4i_gcc*)(uintptr_t)(unsigned)(uintptr_t)lds_ptr,
        0, 0);
#else
    unsigned lds = (unsigned)(uintptr_t)lds_ptr;   // low 32 bits of generic ptr == LDS addr
    asm volatile("global_load_async_to_lds_b128 %0, %1, off"
                 :: "v"(lds), "v"(gptr) : "memory");
#endif
}

__device__ __forceinline__ void wait_asynccnt0() {
#if defined(__has_builtin) && __has_builtin(__builtin_amdgcn_s_wait_asynccnt)
    __builtin_amdgcn_s_wait_asynccnt(0);
#else
    asm volatile("s_wait_asynccnt 0x0" ::: "memory");
#endif
}

// Fused ADMM iteration:  r = K @ Win - q ;  z = soft(r+u,1) ; u += r - z ; Wout = q + z - u
// (FUSED=false variant: plain C = A @ B into fp32 Cout, used for delta = S @ w_final)
template <bool FUSED>
__global__ __launch_bounds__(256) void wmma_gemm_kernel(
    const bf16* __restrict__ A,     // [rowsPadded][KP] bf16, row-major
    const bf16* __restrict__ Bm,    // [KP][NC] bf16, row-major
    const float* __restrict__ q,    // [MROWS][NC]
    float* __restrict__ u,          // [MROWS][NC]
    bf16* __restrict__ Wout,        // [KP][NC]
    float* __restrict__ Cout)       // [SP][NC]
{
    __shared__ __align__(16) bf16 Asm[2][BM * LDSS];
    __shared__ __align__(16) bf16 Bsm[2][BN * LDSS];

    const int n0 = blockIdx.x * BN;
    const int m0 = blockIdx.y * BM;
    const int tid  = threadIdx.x;
    const int lane = tid & 31;
    const int wid  = tid >> 5;
    const int wm = (wid >> 2) * 64;   // wave M offset: 0 / 64
    const int wn = (wid & 3)  * 32;   // wave N offset: 0/32/64/96

    v8f acc[4][2];
    v8f vz = {0.f,0.f,0.f,0.f,0.f,0.f,0.f,0.f};
#pragma unroll
    for (int i = 0; i < 4; ++i)
#pragma unroll
        for (int j = 0; j < 2; ++j) acc[i][j] = vz;

    // A tile: BMxBK row-major copy, global -> LDS (async on CDNA5)
    auto loadA = [&](int buf, int kt) {
        const int k0 = kt * BK;
#pragma unroll
        for (int i = 0; i < 2; ++i) {
            int ch  = tid * 2 + i;        // 512 chunks of 8 bf16
            int row = ch >> 2;
            int kc  = (ch & 3) * 8;
            const bf16* gp = A + (size_t)(m0 + row) * KP + k0 + kc;
#if USE_ASYNC_LDS
            async_copy_b128(&Asm[buf][row * LDSS + kc], gp);
#else
            *(uint4*)(&Asm[buf][row * LDSS + kc]) = *(const uint4*)gp;
#endif
        }
    };
    // B tile: BKxBN, transposed on LDS store -> Bt[n][k]
    auto loadB = [&](int buf, int kt) {
        const int k0 = kt * BK;
#pragma unroll
        for (int i = 0; i < 2; ++i) {
            int ch   = tid * 2 + i;
            int krow = ch >> 4;
            int nc   = (ch & 15) * 8;
            uint4 t = *(const uint4*)(Bm + (size_t)(k0 + krow) * NC + n0 + nc);
            const bf16* e = (const bf16*)&t;
#pragma unroll
            for (int j = 0; j < 8; ++j)
                Bsm[buf][(nc + j) * LDSS + krow] = e[j];
        }
    };

    int buf = 0;
    loadA(0, 0); loadB(0, 0);
#if USE_ASYNC_LDS
    wait_asynccnt0();
#endif
    __syncthreads();

    for (int kt = 0; kt < KSTEPS; ++kt) {
        if (kt + 1 < KSTEPS) { loadA(buf ^ 1, kt + 1); loadB(buf ^ 1, kt + 1); }

        v16bf af[4], bfr[2];
#pragma unroll
        for (int i = 0; i < 4; ++i) af[i]  = load_frag(&Asm[buf][0], wm + i * 16, lane);
#pragma unroll
        for (int j = 0; j < 2; ++j) bfr[j] = load_frag(&Bsm[buf][0], wn + j * 16, lane);

#pragma unroll
        for (int i = 0; i < 4; ++i)
#pragma unroll
            for (int j = 0; j < 2; ++j)
                acc[i][j] = __builtin_amdgcn_wmma_f32_16x16x32_bf16(
                    false, af[i], false, bfr[j], (short)0, acc[i][j], false, false);

#if USE_ASYNC_LDS
        wait_asynccnt0();          // my async A-tile arrived in LDS
#endif
        __syncthreads();           // everyone's A (async) + B (ds_store) tiles visible
        buf ^= 1;
    }

    // epilogue — C/D layout: VGPR v: lanes 0-15 -> M=v, lanes 16-31 -> M=v+8; N=lane%16
    const int r = lane & 15, half = lane >> 4;
#pragma unroll
    for (int i = 0; i < 4; ++i) {
#pragma unroll
        for (int j = 0; j < 2; ++j) {
            const int gn = n0 + wn + j * 16 + r;
#pragma unroll
            for (int v = 0; v < 8; ++v) {
                const int gm = m0 + wm + i * 16 + v + 8 * half;
                const float av = acc[i][j][v];
                if (FUSED) {
                    if (gm < MROWS) {
                        const size_t idx = (size_t)gm * NC + gn;
                        const float qv = q[idx];
                        const float uv = u[idx];
                        const float rr = av - qv;                     // r = Kw - q
                        const float t  = rr + uv;
                        const float z  = copysignf(fmaxf(fabsf(t) - 1.0f, 0.0f), t);
                        const float un = uv + rr - z;
                        u[idx]    = un;
                        Wout[idx] = (bf16)(qv + z - un);              // next w = q + z - u
                    }
                } else {
                    Cout[(size_t)gm * NC + gn] = av;
                }
            }
        }
    }
}

// ---------------- one-time structured setup (tiny, parallel) ----------------
__global__ void build_P(float* __restrict__ P) {
    int idx = blockIdx.x * blockDim.x + threadIdx.x;
    if (idx >= MROWS * NDIM) return;
    int i = idx / NDIM, j = idx % NDIM;
    float v = 0.f;
    if (i < HM)            { int d = j - i;        v = (d >= 0 && d < 25) ? 1.f : 0.f; }
    else if (i < HM + NDIM){ v = (j == i - HM) ? 1.f : 0.f; }                     // REG1*I
    else { int ii = i - (HM + NDIM); v = (j == ii) ? 1.f : ((j == ii + 1) ? -1.f : 0.f); } // REG2*D
    P[idx] = v;
}

__global__ void compute_ptp(const float* __restrict__ P, float* __restrict__ A) {
    int idx = blockIdx.x * blockDim.x + threadIdx.x;
    if (idx >= NDIM * NDIM) return;
    int i = idx / NDIM, j = idx % NDIM;
    float s = 0.f;
    for (int k = 0; k < MROWS; ++k) s += P[k * NDIM + i] * P[k * NDIM + j];
    A[idx] = s;
}

__global__ void compute_scale(const float* __restrict__ A, float* __restrict__ scl) {
    __shared__ float red[512];
    float m = 0.f;
    for (int i = threadIdx.x; i < NDIM; i += 512) {
        float s = 0.f;
        for (int j = 0; j < NDIM; ++j) s += fabsf(A[i * NDIM + j]);
        m = fmaxf(m, s);
    }
    red[threadIdx.x] = m; __syncthreads();
    for (int o = 256; o > 0; o >>= 1) {
        if (threadIdx.x < o) red[threadIdx.x] = fmaxf(red[threadIdx.x], red[threadIdx.x + o]);
        __syncthreads();
    }
    if (threadIdx.x == 0) scl[0] = 1.0f / red[0];   // 1/upper-bound(lambda_max)
}

__global__ void init_X(float* __restrict__ X, const float* __restrict__ scl) {
    int idx = blockIdx.x * blockDim.x + threadIdx.x;
    if (idx >= NDIM * NDIM) return;
    int i = idx / NDIM, j = idx % NDIM;
    X[idx] = (i == j) ? scl[0] : 0.f;
}

__global__ void ns_gemm(float* __restrict__ T, const float* __restrict__ A,
                        const float* __restrict__ X) {   // T = A @ X
    int idx = blockIdx.x * blockDim.x + threadIdx.x;
    if (idx >= NDIM * NDIM) return;
    int i = idx / NDIM, j = idx % NDIM;
    float s = 0.f;
    for (int k = 0; k < NDIM; ++k) s += A[i * NDIM + k] * X[k * NDIM + j];
    T[idx] = s;
}

__global__ void ns_update(float* __restrict__ Xn, const float* __restrict__ X,
                          const float* __restrict__ T) { // Xn = 2X - X @ T
    int idx = blockIdx.x * blockDim.x + threadIdx.x;
    if (idx >= NDIM * NDIM) return;
    int i = idx / NDIM, j = idx % NDIM;
    float s = 0.f;
    for (int k = 0; k < NDIM; ++k) s += X[i * NDIM + k] * T[k * NDIM + j];
    Xn[idx] = 2.f * X[idx] - s;
}

__global__ void compute_PG(const float* __restrict__ P, const float* __restrict__ G,
                           float* __restrict__ PG) {     // PG = P @ G  (G symmetric)
    int idx = blockIdx.x * blockDim.x + threadIdx.x;
    if (idx >= MROWS * NDIM) return;
    int i = idx / NDIM, j = idx % NDIM;
    float s = 0.f;
    for (int k = 0; k < NDIM; ++k) s += P[i * NDIM + k] * G[j * NDIM + k];
    PG[idx] = s;
}

__global__ void compute_Km(const float* __restrict__ PG, const float* __restrict__ P,
                           bf16* __restrict__ Km) {      // Km[MP][KP] = PG @ P^T  (bf16, zero-padded)
    int idx = blockIdx.x * blockDim.x + threadIdx.x;
    if (idx >= MP * KP) return;
    int i = idx / KP, j = idx % KP;
    float s = 0.f;
    if (i < MROWS && j < MROWS)
        for (int k = 0; k < NDIM; ++k) s += PG[i * NDIM + k] * P[j * NDIM + k];
    Km[idx] = (bf16)s;
}

__global__ void compute_S(const float* __restrict__ G, const float* __restrict__ P,
                          bf16* __restrict__ S) {        // S[SP][KP] = G @ P^T (bf16, zero-padded)
    int idx = blockIdx.x * blockDim.x + threadIdx.x;
    if (idx >= SP * KP) return;
    int i = idx / KP, j = idx % KP;
    float s = 0.f;
    if (i < NDIM && j < MROWS)
        for (int k = 0; k < NDIM; ++k) s += G[i * NDIM + k] * P[j * NDIM + k];
    S[idx] = (bf16)s;
}

// q from x; w0 = q (bf16); u = 0; zero-fill both w buffers' padded rows
__global__ void init_data(const float* __restrict__ x, float* __restrict__ q,
                          float* __restrict__ u, bf16* __restrict__ wA,
                          bf16* __restrict__ wB) {
    int idx = blockIdx.x * blockDim.x + threadIdx.x;
    if (idx >= KP * NC) return;
    int k = idx / NC, c = idx % NC;
    float qv = 0.f;
    if (k < HM) {
        int b = c >> 9, e = c & 511;
        qv = x[(size_t)(b * LEN + k + 25) * EMB + e] - x[(size_t)(b * LEN + k) * EMB + e];
    }
    if (k < MROWS) { q[(size_t)k * NC + c] = qv; u[(size_t)k * NC + c] = 0.f; }
    wA[idx] = (bf16)qv;
    wB[idx] = (bf16)0.f;
}

// cumsum delta -> rel / detrend
__global__ void finalize(const float* __restrict__ x, const float* __restrict__ delta,
                         float* __restrict__ out) {
    int c = blockIdx.x * blockDim.x + threadIdx.x;
    if (c >= NC) return;
    int b = c >> 9, e = c & 511;
    for (int l = 0; l < 2; ++l) {
        size_t xi = (size_t)(b * LEN + l) * EMB + e;
        out[xi] = x[xi];
        out[OUTHALF + xi] = 0.f;
    }
    float run = 0.f;
#pragma unroll 2
    for (int k = 0; k < LEN - 2; ++k) {      // delta rows 0..333
        run += delta[(size_t)k * NC + c];
        size_t xi = (size_t)(b * LEN + k + 2) * EMB + e;
        out[xi] = x[xi] - run;
        out[OUTHALF + xi] = run;
    }
}

// ---------------- host launch ----------------
static inline size_t align256(size_t v) { return (v + 255) & ~(size_t)255; }

extern "C" void kernel_launch(void* const* d_in, const int* in_sizes, int n_in,
                              void* d_out, int out_size, void* d_ws, size_t ws_size,
                              hipStream_t stream) {
    (void)in_sizes; (void)n_in; (void)out_size; (void)ws_size;
    const float* x = (const float*)d_in[0];
    float* out = (float*)d_out;

    char* w = (char*)d_ws;
    size_t off = 0;
    auto alloc = [&](size_t bytes) { void* p = w + off; off = align256(off + bytes); return p; };

    float* P    = (float*)alloc((size_t)MROWS * NDIM * 4);
    float* A    = (float*)alloc((size_t)NDIM * NDIM * 4);
    float* Xa   = (float*)alloc((size_t)NDIM * NDIM * 4);
    float* Xb   = (float*)alloc((size_t)NDIM * NDIM * 4);
    float* T    = (float*)alloc((size_t)NDIM * NDIM * 4);
    float* scl  = (float*)alloc(256);
    float* PG   = (float*)alloc((size_t)MROWS * NDIM * 4);
    bf16*  Km   = (bf16*) alloc((size_t)MP * KP * 2);
    bf16*  S    = (bf16*) alloc((size_t)SP * KP * 2);
    float* q    = (float*)alloc((size_t)MROWS * NC * 4);
    float* u    = (float*)alloc((size_t)MROWS * NC * 4);
    bf16*  wA   = (bf16*) alloc((size_t)KP * NC * 2);
    bf16*  wB   = (bf16*) alloc((size_t)KP * NC * 2);
    float* delta= (float*)alloc((size_t)SP * NC * 4);

    const int TB = 256;
    build_P<<<(MROWS * NDIM + TB - 1) / TB, TB, 0, stream>>>(P);
    compute_ptp<<<(NDIM * NDIM + TB - 1) / TB, TB, 0, stream>>>(P, A);
    compute_scale<<<1, 512, 0, stream>>>(A, scl);
    init_X<<<(NDIM * NDIM + TB - 1) / TB, TB, 0, stream>>>(Xa, scl);

    // Newton-Schulz inverse: 16 iterations (cond(PtP) <~ 630, X0 = I/||A||inf)
    float* X = Xa; float* Xn = Xb;
    for (int it = 0; it < 16; ++it) {
        ns_gemm  <<<(NDIM * NDIM + TB - 1) / TB, TB, 0, stream>>>(T, A, X);
        ns_update<<<(NDIM * NDIM + TB - 1) / TB, TB, 0, stream>>>(Xn, X, T);
        float* tmp = X; X = Xn; Xn = tmp;
    }
    float* G = X;

    compute_PG<<<(MROWS * NDIM + TB - 1) / TB, TB, 0, stream>>>(P, G, PG);
    compute_Km<<<(MP * KP + TB - 1) / TB, TB, 0, stream>>>(PG, P, Km);
    compute_S <<<(SP * KP + TB - 1) / TB, TB, 0, stream>>>(G, P, S);

    init_data<<<((size_t)KP * NC + TB - 1) / TB, TB, 0, stream>>>(x, q, u, wA, wB);

    // 50 fused ADMM iterations: one WMMA GEMM + epilogue per iteration (ping-pong w)
    bf16* win = wA; bf16* wout = wB;
    dim3 gAdmm(NC / BN, MP / BM);   // 32 x 8
    for (int it = 0; it < 50; ++it) {
        wmma_gemm_kernel<true><<<gAdmm, TB, 0, stream>>>(Km, win, q, u, wout, nullptr);
        bf16* tmp = win; win = wout; wout = tmp;
    }

    // delta = S @ w_final
    dim3 gFin(NC / BN, SP / BM);    // 32 x 3
    wmma_gemm_kernel<false><<<gFin, TB, 0, stream>>>(S, win, nullptr, nullptr, nullptr, delta);

    finalize<<<NC / TB, TB, 0, stream>>>(x, delta, out);
}
